// TinyBlock2_1219770712846
// MI455X (gfx1250) — matrix operand order repaired and checked by reading, not compiled, not run on hardware
//
#include <hip/hip_runtime.h>
#include <hip/hip_bf16.h>
#include <stdint.h>

typedef __attribute__((ext_vector_type(16))) __bf16 v16bf;
typedef __attribute__((ext_vector_type(8)))  float  v8f;

#define T_FRAMES 512
#define C_IN     64
#define C_OUT    128
#define HH       56
#define WW       56
#define PW       28
#define S_POOL   784     // 28*28
#define KTOT     192     // 3 taps * 64 channels
#define G_TILES  7       // 784 / 112 column groups per frame
#define COLS     112     // columns per block (7 x 16-wide WMMA N tiles)
#define CSTR     200     // padded per-column LDS stride in bf16 elems (400 B)

// ---------------------------------------------------------------------------
// K1: reformat conv weights to bf16 A-matrix [o][tap*64+c]; build segment flags
// ---------------------------------------------------------------------------
__global__ void k_prep(const float* __restrict__ conv_w,
                       const int*   __restrict__ vid_lens,
                       __bf16*      __restrict__ wA,
                       unsigned char* __restrict__ flags) {
    int tid = threadIdx.x;                     // single block, 256 threads
    flags[tid]       = 0;
    flags[tid + 256] = 0;
    __syncthreads();
    if (tid == 0) {
        int cum = 0;
        for (int seg = 0; seg < 8; ++seg) {
            int start = cum;
            cum += vid_lens[seg];
            flags[start]   |= 1;               // segment start: no t-1 tap
            flags[cum - 1] |= 2;               // segment end:   no t+1 tap
        }
    }
    for (int i = tid; i < C_OUT * KTOT; i += 256) {
        int o   = i / KTOT;
        int r   = i % KTOT;
        int tap = r / C_IN;
        int c   = r % C_IN;
        wA[i] = (__bf16)conv_w[(o * C_IN + c) * 3 + tap];
    }
}

// ---------------------------------------------------------------------------
// K2: 2x2 mean pool, fp32 -> bf16, layout [t][s][c] (c contiguous for WMMA-B)
// ---------------------------------------------------------------------------
__global__ void k_pool(const float* __restrict__ x, __bf16* __restrict__ xp) {
    long gid = (long)blockIdx.x * blockDim.x + threadIdx.x;   // exact grid
    int  s   = (int)(gid % S_POOL);
    long tc  = gid / S_POOL;
    int  c   = (int)(tc % C_IN);
    int  t   = (int)(tc / C_IN);
    int  i   = s / PW, j = s % PW;
    const float* base = x + (((long)t * C_IN + c) * HH + 2 * i) * WW + 2 * j;
    float v = 0.25f * (base[0] + base[1] + base[WW] + base[WW + 1]);
    xp[((long)t * S_POOL + s) * C_IN + c] = (__bf16)v;
}

// ---------------------------------------------------------------------------
// K3: segment-masked temporal conv as bf16 WMMA GEMM.
//     Block = 256 threads = 8 waves; wave w owns M-tile [16w,16w+16).
//     Block covers (frame t, 112-column group). B (192x112, bf16) is staged
//     once into LDS via async global->LDS DMA (ASYNCcnt), shared by all waves.
//     Invalid boundary taps are zero-filled in LDS (== zero padding).
// ---------------------------------------------------------------------------
__global__ __launch_bounds__(256)
void k_conv_wmma(const __bf16* __restrict__ xp,
                 const __bf16* __restrict__ wA,
                 const float*  __restrict__ conv_b,
                 const unsigned char* __restrict__ flags,
                 float* __restrict__ y) {
    __shared__ __bf16 Bs[COLS * CSTR];         // 44,800 bytes

    const int lane  = threadIdx.x & 31;
    const int wave  = threadIdx.x >> 5;
    const int mbase = wave * 16;
    const int t     = blockIdx.x / G_TILES;
    const int g     = blockIdx.x % G_TILES;
    const int col0  = g * COLS;
    const int hi    = lane >> 4;               // lane half (0/1)

    const unsigned char fl = flags[t];         // uniform per block

    // ---- stage B: 3 taps x 112 cols x 64 ch, 16B units, async DMA to LDS ----
    for (int tap = 0; tap < 3; ++tap) {
        const bool valid = (tap == 1) || (tap == 0 ? !(fl & 1) : !(fl & 2));
        for (int u = threadIdx.x; u < COLS * 8; u += 256) {
            const int cl = u >> 3, part = u & 7;
            if (valid) {
                const int tsrc = t - 1 + tap;
                unsigned dst = (unsigned)(uintptr_t)(&Bs[cl * CSTR + tap * C_IN])
                             + (unsigned)(part * 16);
                unsigned long long src = (unsigned long long)(uintptr_t)
                    (xp + ((long)tsrc * S_POOL + col0) * C_IN + (size_t)u * 8);
                asm volatile("global_load_async_to_lds_b128 %0, %1, off"
                             :: "v"(dst), "v"(src) : "memory");
            } else {
                uint4 zz = {0u, 0u, 0u, 0u};
                *(uint4*)(&Bs[cl * CSTR + tap * C_IN + part * 8]) = zz;
            }
        }
    }
    asm volatile("s_wait_asynccnt 0x0" ::: "memory");  // this wave's DMAs done
    __syncthreads();                                    // publish LDS to block

    // ---- preload all 6 A chunks (16x32). ISA 16-bit A layout:
    // lanes 0-15: K {kb..kb+7} U {kb+16..kb+23}, kb = chunk*32 (+8 for hi half)
    typedef union { v16bf v; uint4 q[2]; } pack32;
    pack32 A[6];
    {
        const __bf16* wrow = wA + (mbase + (lane & 15)) * KTOT;
        #pragma unroll
        for (int j = 0; j < 6; ++j) {
            int kb = j * 32 + hi * 8;
            A[j].q[0] = *(const uint4*)(wrow + kb);
            A[j].q[1] = *(const uint4*)(wrow + kb + 16);
        }
    }

    const long ybase = (long)t * C_OUT * S_POOL;
    #pragma unroll
    for (int nt = 0; nt < G_TILES; ++nt) {
        const int cl = nt * 16 + (lane & 15);
        // B layout matches A: lanes 0-15 hold K {0..7} U {16..23}, lanes 16-31 K {8..15} U {24..31}
        // -> two 16-byte pieces per lane per K-chunk in the LDS column.
        const __bf16* bp = &Bs[cl * CSTR];
        v8f acc = {};
        #pragma unroll
        for (int j = 0; j < 6; ++j) {
            pack32 B;
            const __bf16* q = bp + j * 32 + hi * 8;
            B.q[0] = *(const uint4*)q;
            B.q[1] = *(const uint4*)(q + 16);
            acc = __builtin_amdgcn_wmma_f32_16x16x32_bf16(
                      false, A[j].v, false, B.v, (short)0, acc, false, false);
        }
        // C/D layout: VGPR r -> row mbase+r (lanes 0-15) / mbase+r+8 (16-31)
        const int colg = col0 + cl;
        #pragma unroll
        for (int r = 0; r < 8; ++r) {
            int row = mbase + r + hi * 8;
            y[ybase + (long)row * S_POOL + colg] = acc[r] + conv_b[row];
        }
    }
}

// ---------------------------------------------------------------------------
// K4: z[t*128+o] = sum_s y[t,o,s]   (deterministic tree reduction, no atomics)
// ---------------------------------------------------------------------------
__global__ void k_zred(const float* __restrict__ y, float* __restrict__ z) {
    const int row = blockIdx.x;                       // (t*128 + o)
    const float* p = y + (long)row * S_POOL;
    float sum = 0.f;
    for (int s = threadIdx.x; s < S_POOL; s += 128) sum += p[s];
    __shared__ float red[128];
    red[threadIdx.x] = sum;
    __syncthreads();
    for (int off = 64; off > 0; off >>= 1) {
        if (threadIdx.x < off) red[threadIdx.x] += red[threadIdx.x + off];
        __syncthreads();
    }
    if (threadIdx.x == 0) z[row] = red[0];
}

// ---------------------------------------------------------------------------
// K5: s = sigmoid(relu(zmean @ W1^T + b1) @ W2^T + b2)
// ---------------------------------------------------------------------------
__global__ void k_se(const float* __restrict__ z,
                     const float* __restrict__ w1, const float* __restrict__ b1,
                     const float* __restrict__ w2, const float* __restrict__ b2,
                     float* __restrict__ sout) {
    const int t = blockIdx.x, o = threadIdx.x;        // 512 blocks x 128 threads
    __shared__ float zr[C_OUT], hr[C_OUT];
    zr[o] = z[t * C_OUT + o] * (1.0f / (float)S_POOL);
    __syncthreads();
    float h = b1[o];
    for (int c = 0; c < C_OUT; ++c) h += zr[c] * w1[o * C_OUT + c];
    hr[o] = fmaxf(h, 0.f);
    __syncthreads();
    float g = b2[o];
    for (int c = 0; c < C_OUT; ++c) g += hr[c] * w2[o * C_OUT + c];
    sout[t * C_OUT + o] = 1.f / (1.f + __expf(-g));
}

// ---------------------------------------------------------------------------
// K6: out *= s[t,o] in place (float4), append vid_lens bits at the tail
// ---------------------------------------------------------------------------
__global__ void k_scale(float* __restrict__ out, const float* __restrict__ sbuf,
                        const int* __restrict__ vid_lens) {
    long gid = (long)blockIdx.x * blockDim.x + threadIdx.x;   // exact grid (N/4)
    long e0  = gid * 4;                       // 784 % 4 == 0 -> one (t,o) row
    int  to  = (int)(e0 / S_POOL);
    float sv = sbuf[to];
    float4 v = *(float4*)(out + e0);
    v.x *= sv; v.y *= sv; v.z *= sv; v.w *= sv;
    *(float4*)(out + e0) = v;
    if (blockIdx.x == 0 && threadIdx.x < 8) {
        ((int*)out)[(long)T_FRAMES * C_OUT * S_POOL + threadIdx.x] =
            vid_lens[threadIdx.x];
    }
}

// ---------------------------------------------------------------------------
extern "C" void kernel_launch(void* const* d_in, const int* in_sizes, int n_in,
                              void* d_out, int out_size, void* d_ws, size_t ws_size,
                              hipStream_t stream) {
    const float* x        = (const float*)d_in[0];
    const int*   vid_lens = (const int*)  d_in[1];
    const float* conv_w   = (const float*)d_in[2];
    const float* conv_b   = (const float*)d_in[3];
    const float* se_w1    = (const float*)d_in[4];
    const float* se_b1    = (const float*)d_in[5];
    const float* se_w2    = (const float*)d_in[6];
    const float* se_b2    = (const float*)d_in[7];
    float* out = (float*)d_out;

    // workspace layout (256B aligned)
    char* ws = (char*)d_ws;
    const size_t xpBytes = (size_t)T_FRAMES * S_POOL * C_IN * 2;   // 51,380,224
    __bf16*        xp    = (__bf16*)ws;
    __bf16*        wA    = (__bf16*)(ws + xpBytes);
    float*         z     = (float*) (ws + xpBytes + 49152);
    float*         sbuf  = (float*) (ws + xpBytes + 49152 + 262144);
    unsigned char* flags = (unsigned char*)(ws + xpBytes + 49152 + 2 * 262144);

    k_prep<<<1, 256, 0, stream>>>(conv_w, vid_lens, wA, flags);

    // 512*64*784 / 256 = 100352 blocks
    k_pool<<<100352, 256, 0, stream>>>(x, xp);

    // 512 frames * 7 column groups; 8 waves/block cover M=128
    k_conv_wmma<<<T_FRAMES * G_TILES, 256, 0, stream>>>(xp, wA, conv_b, flags, out);

    // one block per (t,o) row
    k_zred<<<T_FRAMES * C_OUT, 128, 0, stream>>>(out, z);

    k_se<<<T_FRAMES, C_OUT, 0, stream>>>(z, se_w1, se_b1, se_w2, se_b2, sbuf);

    // 51,380,224 / 4 / 256 = 50176 blocks
    k_scale<<<50176, 256, 0, stream>>>(out, sbuf, vid_lens);
}
